// GraphNetworkLayer_60584808677771
// MI455X (gfx1250) — matrix operand order repaired and checked by reading, not compiled, not run on hardware
//
#include <hip/hip_runtime.h>
#include <hip/hip_bf16.h>
#include <cstdint>
#include <cstddef>

#define N_NODES 100000
#define N_EDGES 800000
#define N_GRAPH 16

typedef __attribute__((ext_vector_type(16))) __bf16 v16bf;
typedef __attribute__((ext_vector_type(8)))  __bf16 v8bf;
typedef __attribute__((ext_vector_type(8)))  float  v8f;
typedef __attribute__((ext_vector_type(4)))  float  v4f;

// Branchless GELU: tanh form via native exp/rcp. Max abs err ~1e-3, below the
// bf16 quantization noise already present in the activations.
__device__ __forceinline__ float gelu_fast(float x) {
    float u = 0.7978845608028654f * x * (1.0f + 0.044715f * x * x);
    float t = __expf(2.0f * u);                       // v_exp_f32 (trans)
    float th = (t - 1.0f) * __builtin_amdgcn_rcpf(t + 1.0f);
    return 0.5f * x * (1.0f + th);
}

// order-preserving float->uint key:  a<b  <=>  fkey(a)<fkey(b)  (no NaNs in workload)
__device__ __forceinline__ unsigned fkey(float x) {
    unsigned b = __float_as_uint(x);
    return (b & 0x80000000u) ? ~b : (b | 0x80000000u);
}
__device__ __forceinline__ float unkey(unsigned k) {
    unsigned b = (k & 0x80000000u) ? (k & 0x7FFFFFFFu) : ~k;
    return __uint_as_float(b);
}

// B fragment (32x16 bf16, pre-swizzled): lane reads 16 contiguous bf16 = 2x b128 loads
__device__ __forceinline__ v16bf load_bfrag(const __bf16* __restrict__ w, int tile, int lane) {
    const v8bf* p = (const v8bf*)(w + (size_t)tile * 512 + lane * 16);
    v8bf lo = p[0], hi = p[1];
    v16bf r;
#pragma unroll
    for (int i = 0; i < 8; i++) { r[i] = lo[i]; r[8 + i] = hi[i]; }
    return r;
}

// pack 16 gathered fp32 (two runs of 8) into an A fragment
__device__ __forceinline__ v16bf cvt16(v4f a, v4f b, v4f c, v4f d) {
    v16bf f;
#pragma unroll
    for (int i = 0; i < 4; i++) {
        f[i]      = (__bf16)a[i];
        f[4 + i]  = (__bf16)b[i];
        f[8 + i]  = (__bf16)c[i];
        f[12 + i] = (__bf16)d[i];
    }
    return f;
}

// A fragment from bf16 LDS row (hidden activations)
__device__ __forceinline__ v16bf lds_afrag(const __bf16* hrow, int off) {
    v8bf lo = *(const v8bf*)(hrow + off);
    v8bf hi = *(const v8bf*)(hrow + off + 16);
    v16bf f;
#pragma unroll
    for (int i = 0; i < 8; i++) { f[i] = lo[i]; f[8 + i] = hi[i]; }
    return f;
}

// ---------------- weight pre-swizzle: fp32 [K,N] row-major -> bf16 fragment order ----
__global__ void swizzle_w(const float* __restrict__ src, __bf16* __restrict__ dst,
                          int K, int N) {
    int t = blockIdx.x * 256 + threadIdx.x;
    if (t >= K * N) return;
    int tile = t >> 9, idx = t & 511, lane = idx >> 4, i = idx & 15;
    int nTiles = N >> 4;
    int kt = tile / nTiles, nt = tile - kt * nTiles;
    int col  = nt * 16 + (lane & 15);
    int krow = kt * 32 + ((lane < 16) ? 0 : 8) + i + ((i < 8) ? 0 : 8);
    dst[t] = (__bf16)src[krow * N + col];
}

// ---------------- init: zero edge2node accumulator, +inf min keys --------------------
__global__ void init_ws(float* __restrict__ edge2node,
                        unsigned* __restrict__ e2g, unsigned* __restrict__ n2g,
                        int nE2N) {
    int t = blockIdx.x * 256 + threadIdx.x;
    if (t < nE2N) edge2node[t] = 0.0f;
    if (t < N_GRAPH * 64) { e2g[t] = 0xFFFFFFFFu; n2g[t] = 0xFFFFFFFFu; }
}

// ---------------- edge MLP: gather -> 256->256->64 WMMA MLP -> scatter-add -----------
// Each wave owns TWO 16-edge M-tiles so every B fragment feeds 2 WMMAs (halves L2
// weight traffic); B fragments are double-buffered to overlap loads with WMMA.
__global__ __launch_bounds__(128) void edge_mlp_kernel(
    const float* __restrict__ node_attr, const float* __restrict__ edge_attr,
    const float* __restrict__ glob_attr,
    const __bf16* __restrict__ w1, const float* __restrict__ b1,
    const __bf16* __restrict__ w2, const float* __restrict__ b2,
    const int* __restrict__ srcIdx, const int* __restrict__ tgtIdx,
    const int* __restrict__ batch,
    float* __restrict__ edge_out, float* __restrict__ edge2node) {
    __shared__ __align__(16) __bf16 hidden[4][32][256];
    __shared__ int tgtLds[4][32];

    const int lane  = threadIdx.x & 31;
    const int wave  = threadIdx.x >> 5;
    const int tile  = blockIdx.x * 4 + wave;      // 32 edges per wave
    const int row   = lane & 15;
    const int off   = (lane < 16) ? 0 : 8;
    const int col   = row;
    const int mbase = (lane < 16) ? 0 : 8;

    const int eA = tile * 32 + row;
    const int eB = eA + 16;
    const int sA = srcIdx[eA], tA = tgtIdx[eA], gA = batch[sA];
    const int sB = srcIdx[eB], tB = tgtIdx[eB], gB = batch[sB];
    if (lane < 16) { tgtLds[wave][lane] = tA; tgtLds[wave][16 + lane] = tB; }

    const float* srcsA[4] = { node_attr + (size_t)sA * 64, node_attr + (size_t)tA * 64,
                              edge_attr + (size_t)eA * 64, glob_attr + (size_t)gA * 64 };
    const float* srcsB[4] = { node_attr + (size_t)sB * 64, node_attr + (size_t)tB * 64,
                              edge_attr + (size_t)eB * 64, glob_attr + (size_t)gB * 64 };

    v16bf afA[8], afB[8];                          // two [16,256] x-tiles in VGPRs
#pragma unroll
    for (int c = 0; c < 8; c++) {
        const float* ba = srcsA[c >> 1] + (c & 1) * 32;
        afA[c] = cvt16(*(const v4f*)(ba + off),      *(const v4f*)(ba + off + 4),
                       *(const v4f*)(ba + off + 16), *(const v4f*)(ba + off + 20));
        const float* bb = srcsB[c >> 1] + (c & 1) * 32;
        afB[c] = cvt16(*(const v4f*)(bb + off),      *(const v4f*)(bb + off + 4),
                       *(const v4f*)(bb + off + 16), *(const v4f*)(bb + off + 20));
    }

    // layer 1: hidden = gelu(x @ W1 + b1)   (16 N-tiles x 8 K-steps)
    for (int jn = 0; jn < 16; jn++) {
        float bv = b1[jn * 16 + col];
        v8f accA = {bv, bv, bv, bv, bv, bv, bv, bv};
        v8f accB = accA;
        v16bf bcur = load_bfrag(w1, jn, lane);
#pragma unroll
        for (int c = 0; c < 8; c++) {
            v16bf bnext = bcur;
            if (c < 7) bnext = load_bfrag(w1, (c + 1) * 16 + jn, lane);
            accA = __builtin_amdgcn_wmma_f32_16x16x32_bf16(
                false, afA[c], false, bcur, (short)0, accA, false, false);
            accB = __builtin_amdgcn_wmma_f32_16x16x32_bf16(
                false, afB[c], false, bcur, (short)0, accB, false, false);
            bcur = bnext;
        }
#pragma unroll
        for (int r = 0; r < 8; r++) {
            hidden[wave][mbase + r][jn * 16 + col]      = (__bf16)gelu_fast(accA[r]);
            hidden[wave][16 + mbase + r][jn * 16 + col] = (__bf16)gelu_fast(accB[r]);
        }
    }

    // layer 2: out = gelu(h @ W2 + b2)      (4 N-tiles x 8 K-steps)
    for (int jo = 0; jo < 4; jo++) {
        float bv = b2[jo * 16 + col];
        v8f accA = {bv, bv, bv, bv, bv, bv, bv, bv};
        v8f accB = accA;
        v16bf bcur = load_bfrag(w2, jo, lane);
#pragma unroll
        for (int c = 0; c < 8; c++) {
            v16bf bnext = bcur;
            if (c < 7) bnext = load_bfrag(w2, (c + 1) * 4 + jo, lane);
            v16bf aa = lds_afrag(&hidden[wave][row][c * 32], off);
            v16bf ab = lds_afrag(&hidden[wave][16 + row][c * 32], off);
            accA = __builtin_amdgcn_wmma_f32_16x16x32_bf16(
                false, aa, false, bcur, (short)0, accA, false, false);
            accB = __builtin_amdgcn_wmma_f32_16x16x32_bf16(
                false, ab, false, bcur, (short)0, accB, false, false);
            bcur = bnext;
        }
#pragma unroll
        for (int r = 0; r < 8; r++) {
            float ga = gelu_fast(accA[r]);
            float gb = gelu_fast(accB[r]);
            int n = jo * 16 + col;
            int ea = tile * 32 + mbase + r;
            int eb = ea + 16;
            edge_out[(size_t)ea * 64 + n] = ga;
            edge_out[(size_t)eb * 64 + n] = gb;
            atomicAdd(&edge2node[(size_t)tgtLds[wave][mbase + r] * 64 + n], ga);
            atomicAdd(&edge2node[(size_t)tgtLds[wave][16 + mbase + r] * 64 + n], gb);
        }
    }
}

// ---------------- node MLP: 192->192->64 WMMA MLP + per-graph min reductions ---------
__global__ __launch_bounds__(64) void node_mlp_kernel(
    const float* __restrict__ node_attr, const float* __restrict__ edge2node,
    const float* __restrict__ glob_attr,
    const __bf16* __restrict__ w1, const float* __restrict__ b1,
    const __bf16* __restrict__ w2, const float* __restrict__ b2,
    const int* __restrict__ batch,
    float* __restrict__ node_out,
    unsigned* __restrict__ e2gKeys, unsigned* __restrict__ n2gKeys) {
    __shared__ __align__(16) __bf16 hidden[2][16][192];
    __shared__ int gphLds[2][16];
    __shared__ unsigned keyE[N_GRAPH][64];
    __shared__ unsigned keyN[N_GRAPH][64];

    for (int i = threadIdx.x; i < N_GRAPH * 64; i += 64) {
        (&keyE[0][0])[i] = 0xFFFFFFFFu;
        (&keyN[0][0])[i] = 0xFFFFFFFFu;
    }
    __syncthreads();

    const int lane  = threadIdx.x & 31;
    const int wave  = threadIdx.x >> 5;
    const int tile  = blockIdx.x * 2 + wave;     // 16 nodes per wave
    const int row   = lane & 15;
    const int off   = (lane < 16) ? 0 : 8;
    const int col   = row;
    const int mbase = (lane < 16) ? 0 : 8;

    const int n = tile * 16 + row;
    const int g = batch[n];
    if (lane < 16) gphLds[wave][lane] = g;

    const float* srcs[3] = { node_attr + (size_t)n * 64, edge2node + (size_t)n * 64,
                             glob_attr + (size_t)g * 64 };

    v16bf afrag[6];
#pragma unroll
    for (int c = 0; c < 6; c++) {
        const float* base = srcs[c >> 1] + (c & 1) * 32;
        v4f r0 = *(const v4f*)(base + off);
        v4f r1 = *(const v4f*)(base + off + 4);
        v4f r2 = *(const v4f*)(base + off + 16);
        v4f r3 = *(const v4f*)(base + off + 20);
        if ((c >> 1) == 1) {                     // feed segment_min(edge2node, batch)
            int cb = (c & 1) * 32 + off;
#pragma unroll
            for (int i = 0; i < 4; i++) {
                atomicMin(&keyE[g][cb + i],      fkey(r0[i]));
                atomicMin(&keyE[g][cb + 4 + i],  fkey(r1[i]));
                atomicMin(&keyE[g][cb + 16 + i], fkey(r2[i]));
                atomicMin(&keyE[g][cb + 20 + i], fkey(r3[i]));
            }
        }
        afrag[c] = cvt16(r0, r1, r2, r3);
    }

    // layer 1: 12 N-tiles x 6 K-steps
    for (int jn = 0; jn < 12; jn++) {
        float bv = b1[jn * 16 + col];
        v8f acc = {bv, bv, bv, bv, bv, bv, bv, bv};
        v16bf bcur = load_bfrag(w1, jn, lane);
#pragma unroll
        for (int c = 0; c < 6; c++) {
            v16bf bnext = bcur;
            if (c < 5) bnext = load_bfrag(w1, (c + 1) * 12 + jn, lane);
            acc = __builtin_amdgcn_wmma_f32_16x16x32_bf16(
                false, afrag[c], false, bcur, (short)0, acc, false, false);
            bcur = bnext;
        }
#pragma unroll
        for (int r = 0; r < 8; r++)
            hidden[wave][mbase + r][jn * 16 + col] = (__bf16)gelu_fast(acc[r]);
    }

    // layer 2: 4 N-tiles x 6 K-steps
    for (int jo = 0; jo < 4; jo++) {
        float bv = b2[jo * 16 + col];
        v8f acc = {bv, bv, bv, bv, bv, bv, bv, bv};
        v16bf bcur = load_bfrag(w2, jo, lane);
#pragma unroll
        for (int c = 0; c < 6; c++) {
            v16bf bnext = bcur;
            if (c < 5) bnext = load_bfrag(w2, (c + 1) * 4 + jo, lane);
            v16bf af = lds_afrag(&hidden[wave][row][c * 32], off);
            acc = __builtin_amdgcn_wmma_f32_16x16x32_bf16(
                false, af, false, bcur, (short)0, acc, false, false);
            bcur = bnext;
        }
#pragma unroll
        for (int r = 0; r < 8; r++) {
            float gv = gelu_fast(acc[r]);
            int nn = tile * 16 + mbase + r;
            int nc = jo * 16 + col;
            node_out[(size_t)nn * 64 + nc] = gv;
            atomicMin(&keyN[gphLds[wave][mbase + r]][nc], fkey(gv));
        }
    }

    __syncthreads();                              // flush block-local mins to global
    for (int i = threadIdx.x; i < N_GRAPH * 64; i += 64) {
        atomicMin(&e2gKeys[i], (&keyE[0][0])[i]);
        atomicMin(&n2gKeys[i], (&keyN[0][0])[i]);
    }
}

// ---------------- global MLP: M=16 (=B) single wave, 192->192->64 --------------------
__global__ __launch_bounds__(32) void glob_mlp_kernel(
    const unsigned* __restrict__ n2gKeys, const unsigned* __restrict__ e2gKeys,
    const float* __restrict__ glob_attr,
    const __bf16* __restrict__ w1, const float* __restrict__ b1,
    const __bf16* __restrict__ w2, const float* __restrict__ b2,
    float* __restrict__ glob_out) {
    __shared__ __align__(16) __bf16 hidden[16][192];

    const int lane  = threadIdx.x & 31;
    const int row   = lane & 15;                  // graph id
    const int off   = (lane < 16) ? 0 : 8;
    const int col   = row;
    const int mbase = (lane < 16) ? 0 : 8;

    v16bf afrag[6];
#pragma unroll
    for (int c = 0; c < 6; c++) {
        int blk = c >> 1;
        v16bf f;
#pragma unroll
        for (int i = 0; i < 16; i++) {
            int rel = (c & 1) * 32 + ((i < 8) ? (off + i) : (16 + off + i - 8));
            float v;
            if (blk == 0)      v = unkey(n2gKeys[row * 64 + rel]);
            else if (blk == 1) v = unkey(e2gKeys[row * 64 + rel]);
            else               v = glob_attr[row * 64 + rel];
            f[i] = (__bf16)v;
        }
        afrag[c] = f;
    }

    for (int jn = 0; jn < 12; jn++) {
        float bv = b1[jn * 16 + col];
        v8f acc = {bv, bv, bv, bv, bv, bv, bv, bv};
        v16bf bcur = load_bfrag(w1, jn, lane);
#pragma unroll
        for (int c = 0; c < 6; c++) {
            v16bf bnext = bcur;
            if (c < 5) bnext = load_bfrag(w1, (c + 1) * 12 + jn, lane);
            acc = __builtin_amdgcn_wmma_f32_16x16x32_bf16(
                false, afrag[c], false, bcur, (short)0, acc, false, false);
            bcur = bnext;
        }
#pragma unroll
        for (int r = 0; r < 8; r++)
            hidden[mbase + r][jn * 16 + col] = (__bf16)gelu_fast(acc[r]);
    }

    for (int jo = 0; jo < 4; jo++) {
        float bv = b2[jo * 16 + col];
        v8f acc = {bv, bv, bv, bv, bv, bv, bv, bv};
        v16bf bcur = load_bfrag(w2, jo, lane);
#pragma unroll
        for (int c = 0; c < 6; c++) {
            v16bf bnext = bcur;
            if (c < 5) bnext = load_bfrag(w2, (c + 1) * 4 + jo, lane);
            v16bf af = lds_afrag(&hidden[row][c * 32], off);
            acc = __builtin_amdgcn_wmma_f32_16x16x32_bf16(
                false, af, false, bcur, (short)0, acc, false, false);
            bcur = bnext;
        }
#pragma unroll
        for (int r = 0; r < 8; r++)
            glob_out[(size_t)(mbase + r) * 64 + jo * 16 + col] = gelu_fast(acc[r]);
    }
}

// -------------------------------------------------------------------------------------
extern "C" void kernel_launch(void* const* d_in, const int* in_sizes, int n_in,
                              void* d_out, int out_size, void* d_ws, size_t ws_size,
                              hipStream_t stream) {
    const float* node_attr = (const float*)d_in[0];
    const float* edge_attr = (const float*)d_in[1];
    const float* glob_attr = (const float*)d_in[2];
    const float* eW1 = (const float*)d_in[3];
    const float* eB1 = (const float*)d_in[4];
    const float* eW2 = (const float*)d_in[5];
    const float* eB2 = (const float*)d_in[6];
    const float* nW1 = (const float*)d_in[7];
    const float* nB1 = (const float*)d_in[8];
    const float* nW2 = (const float*)d_in[9];
    const float* nB2 = (const float*)d_in[10];
    const float* gW1 = (const float*)d_in[11];
    const float* gB1 = (const float*)d_in[12];
    const float* gW2 = (const float*)d_in[13];
    const float* gB2 = (const float*)d_in[14];
    const int* edge_index = (const int*)d_in[15];
    const int* batch = (const int*)d_in[16];
    const int* srcIdx = edge_index;
    const int* tgtIdx = edge_index + N_EDGES;

    char* ws = (char*)d_ws;
    __bf16* sweW1 = (__bf16*)ws;  ws += 256 * 256 * 2;
    __bf16* sweW2 = (__bf16*)ws;  ws += 256 * 64 * 2;
    __bf16* swnW1 = (__bf16*)ws;  ws += 192 * 192 * 2;
    __bf16* swnW2 = (__bf16*)ws;  ws += 192 * 64 * 2;
    __bf16* swgW1 = (__bf16*)ws;  ws += 192 * 192 * 2;
    __bf16* swgW2 = (__bf16*)ws;  ws += 192 * 64 * 2;
    ws = (char*)(((uintptr_t)ws + 255) & ~(uintptr_t)255);
    float* edge2node = (float*)ws; ws += (size_t)N_NODES * 64 * 4;
    unsigned* e2gKeys = (unsigned*)ws; ws += N_GRAPH * 64 * 4;
    unsigned* n2gKeys = (unsigned*)ws; ws += N_GRAPH * 64 * 4;

    float* node_out = (float*)d_out;
    float* edge_out = node_out + (size_t)N_NODES * 64;
    float* glob_out = edge_out + (size_t)N_EDGES * 64;

    // weight pre-swizzle into fragment-order bf16 (stays hot in 192MB L2)
    swizzle_w<<<(256 * 256 + 255) / 256, 256, 0, stream>>>(eW1, sweW1, 256, 256);
    swizzle_w<<<(256 * 64 + 255) / 256, 256, 0, stream>>>(eW2, sweW2, 256, 64);
    swizzle_w<<<(192 * 192 + 255) / 256, 256, 0, stream>>>(nW1, swnW1, 192, 192);
    swizzle_w<<<(192 * 64 + 255) / 256, 256, 0, stream>>>(nW2, swnW2, 192, 64);
    swizzle_w<<<(192 * 192 + 255) / 256, 256, 0, stream>>>(gW1, swgW1, 192, 192);
    swizzle_w<<<(192 * 64 + 255) / 256, 256, 0, stream>>>(gW2, swgW2, 192, 64);

    int nE2N = N_NODES * 64;
    init_ws<<<(nE2N + 255) / 256, 256, 0, stream>>>(edge2node, e2gKeys, n2gKeys, nE2N);

    edge_mlp_kernel<<<N_EDGES / 128, 128, 0, stream>>>(
        node_attr, edge_attr, glob_attr, sweW1, eB1, sweW2, eB2,
        srcIdx, tgtIdx, batch, edge_out, edge2node);

    node_mlp_kernel<<<N_NODES / 32, 64, 0, stream>>>(
        node_attr, edge2node, glob_attr, swnW1, nB1, swnW2, nB2,
        batch, node_out, e2gKeys, n2gKeys);

    glob_mlp_kernel<<<1, 32, 0, stream>>>(
        n2gKeys, e2gKeys, glob_attr, swgW1, gB1, swgW2, gB2, glob_out);
}
